// Resonator_25091198943365
// MI455X (gfx1250) — 16-bit and 8-bit matrix operand order repaired and checked by reading, compiled, not run on hardware
//
#include <hip/hip_runtime.h>
#include <stdint.h>

typedef __attribute__((ext_vector_type(16))) int    v16i;
typedef __attribute__((ext_vector_type(8)))  float  v8f;
typedef __attribute__((ext_vector_type(16))) __bf16 v16bf;

#define B_ 128
#define F_ 4
#define V_ 64
#define D_ 16384
#define ITERS 10

#define P1 ((unsigned char)0x38)
#define N1 ((unsigned char)0xB8)

#if defined(__gfx1250__) && __has_builtin(__builtin_amdgcn_global_load_async_to_lds_b128)
#define HAVE_ASYNC_LDS 1
typedef int v4i_vs __attribute__((vector_size(16)));
typedef __attribute__((address_space(1))) v4i_vs* async_gptr;
typedef __attribute__((address_space(3))) v4i_vs* async_lptr;
#endif

__global__ void k_init_cb(const float* __restrict__ cb,
                          unsigned char* __restrict__ cb8,
                          unsigned short* __restrict__ cbT) {
  int i = blockIdx.x * blockDim.x + threadIdx.x;
  if (i >= F_ * V_ * D_) return;
  int f = i / (V_ * D_);
  int vd = i % (V_ * D_);
  int v = vd / D_;
  int d = vd % D_;
  unsigned s = (cb[i] < 0.0f) ? 1u : 0u;
  cb8[i] = s ? N1 : P1;
  cbT[((size_t)f * D_ + d) * V_ + v] = s ? 0xBF80u : 0x3F80u;
}

__global__ void k_init_est(const float* __restrict__ inp,
                           const float* __restrict__ est0,
                           unsigned char* __restrict__ inp8,
                           unsigned char* __restrict__ est8,
                           unsigned char* __restrict__ nst8) {
  int i = blockIdx.x * blockDim.x + threadIdx.x;
  if (i >= B_ * D_) return;
  int b = i / D_;
  int d = i % D_;
  unsigned char si = (inp[i] < 0.0f) ? N1 : P1;
  inp8[i] = si;
  unsigned char e[F_];
  unsigned char x = si;
  #pragma unroll
  for (int f = 0; f < F_; ++f) {
    e[f] = (est0[((size_t)b * F_ + f) * D_ + d] < 0.0f) ? N1 : P1;
    est8[((size_t)b * F_ + f) * D_ + d] = e[f];
    x ^= e[f];
  }
  #pragma unroll
  for (int f = 0; f < F_; ++f)
    nst8[((size_t)b * F_ + f) * D_ + d] = (unsigned char)((x ^ e[f]) | P1);
}

__global__ void k_zero(float* __restrict__ p, int n) {
  int i = blockIdx.x * blockDim.x + threadIdx.x;
  if (i < n) p[i] = 0.0f;
}

__global__ void k_sim(const unsigned char* __restrict__ src,
                      const unsigned char* __restrict__ cb8,
                      float* __restrict__ sim) {
  const int bt = blockIdx.x;
  const int f  = blockIdx.y;
  const int z  = blockIdx.z;
  const int w    = threadIdx.x >> 5;
  const int lane = threadIdx.x & 31;
  const int l15  = lane & 15;
  const int kh   = lane >> 4;

  const unsigned char* arow  = src + (((size_t)(bt * 16 + l15) * F_ + f) * D_);
  const unsigned char* bbase = cb8 + (size_t)f * V_ * D_;

  const v8f zero8 = {0.f, 0.f, 0.f, 0.f, 0.f, 0.f, 0.f, 0.f};
  v8f c[4];
  #pragma unroll
  for (int t = 0; t < 4; ++t) c[t] = zero8;

  const int k0 = z * 2048 + w * 512;
  #pragma unroll
  for (int ks = 0; ks < 4; ++ks) {
    const int k = k0 + ks * 128;
    __builtin_prefetch(arow + k + 128, 0, 1);
    v16i a;
    #pragma unroll
    for (int p = 0; p < 8; ++p) {
      const int j = p >> 2, g = p & 3;
      const uint2 u = *reinterpret_cast<const uint2*>(arow + k + kh * 16 + (p >> 1) * 32 + (p & 1) * 8);
      a[2 * p]     = (int)u.x;
      a[2 * p + 1] = (int)u.y;
    }
    #pragma unroll
    for (int t = 0; t < 4; ++t) {
      const unsigned char* bp = bbase + (size_t)(t * 16 + l15) * D_ + k + kh * 16;
      v16i bm;
      #pragma unroll
      for (int q = 0; q < 4; ++q) {
        const uint4 u = *reinterpret_cast<const uint4*>(bp + q * 32);
        bm[4 * q] = (int)u.x; bm[4 * q + 1] = (int)u.y;
        bm[4 * q + 2] = (int)u.z; bm[4 * q + 3] = (int)u.w;
      }
      c[t] = __builtin_amdgcn_wmma_f32_16x16x128_fp8_fp8(a, bm, (short)0, c[t], false, false);
    }
  }

  #pragma unroll
  for (int t = 0; t < 4; ++t) {
    #pragma unroll
    for (int r = 0; r < 8; ++r) {
      const int brow = bt * 16 + kh * 8 + r;
      const int v    = t * 16 + l15;
      atomicAdd(&sim[((size_t)brow * F_ + f) * V_ + v], c[t][r]);
    }
  }
}

__global__ void k_update(const float* __restrict__ sim,
                         const unsigned short* __restrict__ cbT,
                         const unsigned char* __restrict__ inp8,
                         unsigned char* __restrict__ est8,
                         unsigned char* __restrict__ nst8) {
  __shared__ float lsim[16 * F_ * V_];
  const int bt  = blockIdx.x;
  const int dc  = blockIdx.y;
  const int tid = threadIdx.x;

  {
    const float* gsrc = sim + (size_t)bt * 16 * F_ * V_;
#ifdef HAVE_ASYNC_LDS
    for (int i = tid; i < 1024; i += 128) {
      __builtin_amdgcn_global_load_async_to_lds_b128(
          (async_gptr)(gsrc + 4 * i),
          (async_lptr)(lsim + 4 * i),
          0, 0);
    }
#if __has_builtin(__builtin_amdgcn_s_wait_asynccnt)
    __builtin_amdgcn_s_wait_asynccnt(0);
#else
    asm volatile("s_wait_asynccnt 0x0" ::: "memory");
#endif
#else
    const float4* g = reinterpret_cast<const float4*>(gsrc);
    float4* l = reinterpret_cast<float4*>(lsim);
    for (int i = tid; i < 1024; i += 128) l[i] = g[i];
#endif
  }
  __syncthreads();

  const int w = tid >> 5, lane = tid & 31, l15 = lane & 15, kh = lane >> 4;
  const int dwave = dc * 512 + w * 128;
  const v8f zero8 = {0.f, 0.f, 0.f, 0.f, 0.f, 0.f, 0.f, 0.f};

  union BU { uint4 q[2]; v16bf v; };

  unsigned int mask[8] = {0, 0, 0, 0, 0, 0, 0, 0};

  #pragma unroll
  for (int f = 0; f < F_; ++f) {
    v16bf Ah[2], Al[2];
    #pragma unroll
    for (int ks = 0; ks < 2; ++ks) {
      #pragma unroll
      for (int i = 0; i < 16; ++i) {
        const int K = kh * 16 + i;
        const float val = lsim[(l15 * F_ + f) * V_ + ks * 32 + K];
        const __bf16 h = (__bf16)val;
        Ah[ks][i] = h;
        Al[ks][i] = (__bf16)(val - (float)h);
      }
    }
    #pragma unroll
    for (int nt = 0; nt < 8; ++nt) {
      const int dcol = dwave + nt * 16 + l15;
      v8f acc = zero8;
      #pragma unroll
      for (int ks = 0; ks < 2; ++ks) {
        BU bu;
        const uint4* bp = reinterpret_cast<const uint4*>(
            cbT + ((size_t)f * D_ + dcol) * V_ + ks * 32 + kh * 16);
        bu.q[0] = bp[0];
        bu.q[1] = bp[1];
        acc = __builtin_amdgcn_wmma_f32_16x16x32_bf16(false, Ah[ks], false, bu.v,
                                                      (short)0, acc, false, false);
        acc = __builtin_amdgcn_wmma_f32_16x16x32_bf16(false, Al[ks], false, bu.v,
                                                      (short)0, acc, false, false);
      }
      #pragma unroll
      for (int r = 0; r < 8; ++r) {
        const unsigned s = (acc[r] < 0.0f) ? 1u : 0u;
        mask[nt] |= s << (f * 8 + r);
        est8[((size_t)(bt * 16 + kh * 8 + r) * F_ + f) * D_ + dcol] = s ? N1 : P1;
      }
    }
  }

  #pragma unroll
  for (int nt = 0; nt < 8; ++nt) {
    const unsigned m = mask[nt];
    #pragma unroll
    for (int r = 0; r < 8; ++r) {
      const int brow = bt * 16 + kh * 8 + r;
      const int dcol = dwave + nt * 16 + l15;
      const unsigned si = (unsigned)(inp8[(size_t)brow * D_ + dcol] >> 7) & 1u;
      const unsigned sall = ((m >> r) ^ (m >> (8 + r)) ^ (m >> (16 + r)) ^ (m >> (24 + r)) ^ si) & 1u;
      #pragma unroll
      for (int f = 0; f < F_; ++f) {
        const unsigned sf = (m >> (f * 8 + r)) & 1u;
        nst8[((size_t)brow * F_ + f) * D_ + dcol] =
            (unsigned char)(((sall ^ sf) << 7) | P1);
      }
    }
  }
}

__global__ void k_argmax(const float* __restrict__ sim, int* __restrict__ out) {
  int i = blockIdx.x * blockDim.x + threadIdx.x;
  if (i < B_ * F_) {
    const float* row = sim + (size_t)i * V_;
    float best = -1.0f;
    int bi = 0;
    for (int v = 0; v < V_; ++v) {
      float a = fabsf(row[v]);
      if (a > best) { best = a; bi = v; }
    }
    out[i] = bi;
  }
  if (i == 0) out[B_ * F_] = ITERS - 1;
}

extern "C" void kernel_launch(void* const* d_in, const int* in_sizes, int n_in,
                              void* d_out, int out_size, void* d_ws, size_t ws_size,
                              hipStream_t stream) {
  (void)in_sizes; (void)n_in; (void)out_size; (void)ws_size;
  const float* inp  = (const float*)d_in[0];
  const float* est0 = (const float*)d_in[1];
  const float* cb   = (const float*)d_in[2];

  char* ws = (char*)d_ws;
  unsigned char*  cb8  = (unsigned char*) (ws);
  unsigned short* cbT  = (unsigned short*)(ws +  4u * 1024 * 1024);
  unsigned char*  est8 = (unsigned char*) (ws + 12u * 1024 * 1024);
  unsigned char*  nst8 = (unsigned char*) (ws + 20u * 1024 * 1024);
  unsigned char*  inp8 = (unsigned char*) (ws + 28u * 1024 * 1024);
  float*          simb = (float*)         (ws + 30u * 1024 * 1024);

  k_init_cb <<<(F_ * V_ * D_) / 256, 256, 0, stream>>>(cb, cb8, cbT);
  k_init_est<<<(B_ * D_)      / 256, 256, 0, stream>>>(inp, est0, inp8, est8, nst8);

  for (int it = 0; it < ITERS; ++it) {
    k_zero  <<<(B_ * F_ * V_) / 256, 256, 0, stream>>>(simb, B_ * F_ * V_);
    k_sim   <<<dim3(8, 4, 8), 128, 0, stream>>>(nst8, cb8, simb);
    k_update<<<dim3(8, 32),   128, 0, stream>>>(simb, cbT, inp8, est8, nst8);
  }

  k_zero  <<<(B_ * F_ * V_) / 256, 256, 0, stream>>>(simb, B_ * F_ * V_);
  k_sim   <<<dim3(8, 4, 8), 128, 0, stream>>>(est8, cb8, simb);
  k_argmax<<<2, 256, 0, stream>>>(simb, (int*)d_out);
}
